// TAGCN_88192858456069
// MI455X (gfx1250) — matrix operand order repaired and checked by reading, not compiled, not run on hardware
//
#include <hip/hip_runtime.h>
#include <hip/hip_bf16.h>

// ---------------------------------------------------------------------------
// TAGCN (K=2, 3 layers) for MI455X / gfx1250.
//  - SpMM hops: edge-parallel float4 gather + hw f32 atomics (L2-resident).
//  - Weights: one-shot transpose+convert to bf16 [dn, 768] so the GEMM's
//    B fragments are single contiguous 32B loads.
//  - Dense projections: v_wmma_f32_16x16x32_bf16, 32x32 tile per wave
//    (2x2 subtiles, 4 accumulators), native f32->bf16 converts for A,
//    bias + leaky-relu fused epilogue.
// ---------------------------------------------------------------------------

typedef __attribute__((ext_vector_type(16))) __bf16 v16bf;
typedef __attribute__((ext_vector_type(8)))  float  v8f;

#define N_NODES 100000
#define N_EDGES 3200000
#define DFEAT   256
#define KTOT    (3 * DFEAT)       // 768
#define NEG_SLOPE 0.01f

// ---------------------------------------------------------------------------
// SpMM: y[dst] += w * x[src].  One wave (32 lanes) per edge, each lane moves
// two float4 chunks of the 256-wide feature row. 8 edges per 256-thread block.
// ---------------------------------------------------------------------------
__global__ __launch_bounds__(256) void tag_spmm(
    const int*   __restrict__ esrc,
    const int*   __restrict__ edst,
    const float* __restrict__ ew,
    const float* __restrict__ x,
    float*       __restrict__ y)
{
    const int e = blockIdx.x * 8 + (threadIdx.x >> 5);
    if (e >= N_EDGES) return;
    const int lane = threadIdx.x & 31;

    const int   s  = esrc[e];
    const int   d  = edst[e];
    const float wt = ew[e];

    const float4* __restrict__ xs = (const float4*)(x + (size_t)s * DFEAT);
    float*        __restrict__ yd = y + (size_t)d * DFEAT;

#pragma unroll
    for (int c = lane; c < DFEAT / 4; c += 32) {
        float4 v = xs[c];
        float* p = yd + 4 * c;
        __hip_atomic_fetch_add(p + 0, wt * v.x, __ATOMIC_RELAXED, __HIP_MEMORY_SCOPE_AGENT);
        __hip_atomic_fetch_add(p + 1, wt * v.y, __ATOMIC_RELAXED, __HIP_MEMORY_SCOPE_AGENT);
        __hip_atomic_fetch_add(p + 2, wt * v.z, __ATOMIC_RELAXED, __HIP_MEMORY_SCOPE_AGENT);
        __hip_atomic_fetch_add(p + 3, wt * v.w, __ATOMIC_RELAXED, __HIP_MEMORY_SCOPE_AGENT);
    }
}

// ---------------------------------------------------------------------------
// Weight transpose + f32->bf16: W[KTOT, dn] -> Wt[dn, KTOT].  Tiny (<=768 KB),
// runs once per layer; coalesced reads, strided bf16 writes.
// ---------------------------------------------------------------------------
__global__ __launch_bounds__(256) void tag_wconv(
    const float* __restrict__ W, __bf16* __restrict__ Wt, int dn)
{
    const int idx = blockIdx.x * 256 + threadIdx.x;
    if (idx >= KTOT * dn) return;
    const int k = idx / dn;
    const int n = idx - k * dn;
    Wt[(size_t)n * KTOT + k] = (__bf16)W[idx];
}

// ---------------------------------------------------------------------------
// Dense projection of the hop-concat:  out = leaky([f0|f1|f2] @ W + b).
// One wave computes a 32x32 output tile = 2x2 grid of 16x16 WMMA tiles.
// K = 3 * 256, stepped 32 at a time through v_wmma_f32_16x16x32_bf16.
// VGPR layouts used here (contiguous k, same order for A and B):
//   A (16x32 bf16): lane m=l&15, half=l>>4; elem e -> K=half*16+e
//   B (32x16 bf16): N=l&15, elem e -> K=half*16+e  (contiguous in Wt!)
//   C/D (16x16 f32): c[v] -> row (half*8+v), col (l&15)
// ---------------------------------------------------------------------------
__global__ __launch_bounds__(32) void tag_gemm(
    const float*  __restrict__ f0,
    const float*  __restrict__ f1,
    const float*  __restrict__ f2,
    const __bf16* __restrict__ Wt,    // [dn, KTOT] bf16 (transposed)
    const float*  __restrict__ bias,  // [dn]
    float*        __restrict__ out,   // [N_NODES, dn]
    int dn, int act)
{
    const int lane = threadIdx.x;       // 0..31, full wave (EXEC all 1s)
    const int m0   = blockIdx.x * 32;   // node tile (N_NODES/32 = 3125 exact)
    const int n0   = blockIdx.y * 32;   // output-feature tile
    const int l15  = lane & 15;
    const int half = lane >> 4;

    v8f acc00 = {}, acc01 = {}, acc10 = {}, acc11 = {};

    // Per-lane base pointers
    const size_t arow0 = (size_t)(m0 + l15) * DFEAT;        // M-subtile 0
    const size_t arow1 = (size_t)(m0 + 16 + l15) * DFEAT;   // M-subtile 1
    const __bf16* __restrict__ wt0 = Wt + (size_t)(n0 + l15)      * KTOT + half * 16;
    const __bf16* __restrict__ wt1 = Wt + (size_t)(n0 + 16 + l15) * KTOT + half * 16;

    for (int s = 0; s < 3; ++s) {
        const float* __restrict__ src = (s == 0) ? f0 : (s == 1) ? f1 : f2;
        const int kbase = s * DFEAT;

        for (int k0 = 0; k0 < DFEAT; k0 += 32) {
            const int ka = k0 + half * 16;

            // ---- A tiles: two contiguous runs of 8 f32 each, native cvt
            v16bf a0, a1;
            {
                const float4* p = (const float4*)(src + arow0 + ka);
                const float4* q = (const float4*)(src + arow0 + ka + 8);
                float4 r0 = p[0], r1 = p[1], r2 = q[0], r3 = q[1];
                a0[0]=(__bf16)r0.x; a0[1]=(__bf16)r0.y; a0[2]=(__bf16)r0.z; a0[3]=(__bf16)r0.w;
                a0[4]=(__bf16)r1.x; a0[5]=(__bf16)r1.y; a0[6]=(__bf16)r1.z; a0[7]=(__bf16)r1.w;
                a0[8]=(__bf16)r2.x; a0[9]=(__bf16)r2.y; a0[10]=(__bf16)r2.z; a0[11]=(__bf16)r2.w;
                a0[12]=(__bf16)r3.x; a0[13]=(__bf16)r3.y; a0[14]=(__bf16)r3.z; a0[15]=(__bf16)r3.w;
            }
            {
                const float4* p = (const float4*)(src + arow1 + ka);
                const float4* q = (const float4*)(src + arow1 + ka + 8);
                float4 r0 = p[0], r1 = p[1], r2 = q[0], r3 = q[1];
                a1[0]=(__bf16)r0.x; a1[1]=(__bf16)r0.y; a1[2]=(__bf16)r0.z; a1[3]=(__bf16)r0.w;
                a1[4]=(__bf16)r1.x; a1[5]=(__bf16)r1.y; a1[6]=(__bf16)r1.z; a1[7]=(__bf16)r1.w;
                a1[8]=(__bf16)r2.x; a1[9]=(__bf16)r2.y; a1[10]=(__bf16)r2.z; a1[11]=(__bf16)r2.w;
                a1[12]=(__bf16)r3.x; a1[13]=(__bf16)r3.y; a1[14]=(__bf16)r3.z; a1[15]=(__bf16)r3.w;
            }

            // ---- B tiles: single contiguous 32B bf16 load per lane per tile
            v16bf b0 = *(const v16bf*)(wt0 + kbase + k0);
            v16bf b1 = *(const v16bf*)(wt1 + kbase + k0);

            acc00 = __builtin_amdgcn_wmma_f32_16x16x32_bf16(false, a0, false, b0, (short)0, acc00, false, false);
            acc01 = __builtin_amdgcn_wmma_f32_16x16x32_bf16(false, a0, false, b1, (short)0, acc01, false, false);
            acc10 = __builtin_amdgcn_wmma_f32_16x16x32_bf16(false, a1, false, b0, (short)0, acc10, false, false);
            acc11 = __builtin_amdgcn_wmma_f32_16x16x32_bf16(false, a1, false, b1, (short)0, acc11, false, false);
        }
    }

    // ---- Epilogue: bias + leaky-relu, scatter 2x2 subtiles
    const float bv0 = bias[n0 + l15];
    const float bv1 = bias[n0 + 16 + l15];
#pragma unroll
    for (int v = 0; v < 8; ++v) {
        const int mA = m0 + half * 8 + v;        // M-subtile 0 row
        const int mB = mA + 16;                  // M-subtile 1 row
        float r00 = acc00[v] + bv0;
        float r01 = acc01[v] + bv1;
        float r10 = acc10[v] + bv0;
        float r11 = acc11[v] + bv1;
        if (act) {
            r00 = (r00 > 0.0f) ? r00 : r00 * NEG_SLOPE;
            r01 = (r01 > 0.0f) ? r01 : r01 * NEG_SLOPE;
            r10 = (r10 > 0.0f) ? r10 : r10 * NEG_SLOPE;
            r11 = (r11 > 0.0f) ? r11 : r11 * NEG_SLOPE;
        }
        out[(size_t)mA * dn + n0 + l15]      = r00;
        out[(size_t)mA * dn + n0 + 16 + l15] = r01;
        out[(size_t)mB * dn + n0 + l15]      = r10;
        out[(size_t)mB * dn + n0 + 16 + l15] = r11;
    }
}

// ---------------------------------------------------------------------------
// Host orchestration: 3 TAGConv layers, each = 2 SpMM hops + fused GEMM.
// ---------------------------------------------------------------------------
extern "C" void kernel_launch(void* const* d_in, const int* in_sizes, int n_in,
                              void* d_out, int out_size, void* d_ws, size_t ws_size,
                              hipStream_t stream) {
    const float* x    = (const float*)d_in[0];
    const int*   esrc = (const int*)  d_in[1];
    const int*   edst = (const int*)  d_in[2];
    const float* ew   = (const float*)d_in[3];
    const float* W0   = (const float*)d_in[4];
    const float* b0   = (const float*)d_in[5];
    const float* W1   = (const float*)d_in[6];
    const float* b1   = (const float*)d_in[7];
    const float* W2   = (const float*)d_in[8];
    const float* b2   = (const float*)d_in[9];
    float* out = (float*)d_out;

    const size_t featBytes = (size_t)N_NODES * DFEAT * sizeof(float);
    char* ws = (char*)d_ws;
    float* hop1 = (float*)(ws + 0 * featBytes);
    float* hop2 = (float*)(ws + 1 * featBytes);
    float* h0   = (float*)(ws + 2 * featBytes);   // layer-0 output
    float* h1   = (float*)(ws + 3 * featBytes);   // layer-1 output
    __bf16* W0t = (__bf16*)(ws + 4 * featBytes);              // [256, 768]
    __bf16* W1t = W0t + (size_t)KTOT * 256;                   // [256, 768]
    __bf16* W2t = W1t + (size_t)KTOT * 256;                   // [ 64, 768]

    // One-shot weight transpose + bf16 convert (tiny, L2-resident)
    tag_wconv<<<(KTOT * 256 + 255) / 256, 256, 0, stream>>>(W0, W0t, 256);
    tag_wconv<<<(KTOT * 256 + 255) / 256, 256, 0, stream>>>(W1, W1t, 256);
    tag_wconv<<<(KTOT *  64 + 255) / 256, 256, 0, stream>>>(W2, W2t,  64);

    const dim3 sgrid((N_EDGES + 7) / 8), sblk(256);

    auto layer = [&](const float* feat, const __bf16* Wt, const float* bias,
                     float* o, int dn, int act) {
        hipMemsetAsync(hop1, 0, featBytes, stream);
        tag_spmm<<<sgrid, sblk, 0, stream>>>(esrc, edst, ew, feat, hop1);
        hipMemsetAsync(hop2, 0, featBytes, stream);
        tag_spmm<<<sgrid, sblk, 0, stream>>>(esrc, edst, ew, hop1, hop2);
        dim3 g(N_NODES / 32, dn / 32);   // 3125 x (dn/32), exact tiling
        tag_gemm<<<g, 32, 0, stream>>>(feat, hop1, hop2, Wt, bias, o, dn, act);
    };

    layer(x,  W0t, b0, h0,  256, 1);
    layer(h0, W1t, b1, h1,  256, 1);
    layer(h1, W2t, b2, out,  64, 0);
}
